// Seq2SeqTransformer_40209483825525
// MI455X (gfx1250) — matrix operand order repaired and checked by reading, not compiled, not run on hardware
//
#include <hip/hip_runtime.h>
#include <hip/hip_bf16.h>
#include <math.h>

typedef _Float16 v16h __attribute__((ext_vector_type(16)));
typedef float    v8f  __attribute__((ext_vector_type(8)));

namespace {
constexpr int kB = 8, kS = 24, kE = 512, kH = 8, kD = 64, kF = 2048;
constexpr int kLE = 3, kLD = 3;
constexpr int kLbuf = 25, kMaxLen = 24, kVT = 32000, kStart = 1;
constexpr int kTenc = kS * kB;     // 192
constexpr int kTdec = kLbuf * kB;  // 200
}

// ---------------------------------------------------------------------------
// fp32 [K, Nfull] sub-block (cols c0..c0+Nd) -> f16 transposed [Nd, Kd]
// ---------------------------------------------------------------------------
__global__ void cvtT_kernel(const float* __restrict__ w, int Nfull, int c0,
                            int Kd, int Nd, _Float16* __restrict__ wT) {
  int idx = blockIdx.x * blockDim.x + threadIdx.x;
  if (idx >= Nd * Kd) return;
  int n = idx / Kd, kk = idx % Kd;
  wT[(size_t)n * Kd + kk] = (_Float16)w[(size_t)kk * Nfull + c0 + n];
}

// ---------------------------------------------------------------------------
// WMMA GEMM: C[M,N] = A[M,K](fp32) * WT[N,K](f16, pre-transposed) + bias
// One wave per 16x32 output tile: one A fragment feeds two B fragments and
// two v_wmma_f32_16x16x32_f16 per 32-wide K chunk.
// A fragment (16-bit A 16x32): lane row = lane&15;
//   halves h -> K = kc + hi*16 + h (contiguous, matching B).
// B fragment: lane col = lane&15; halves h -> K = kc + hi*16 + h (contiguous).
// C: acc[r] -> row tileM*16 + hi*8 + r, col = lane&15.
// Loads are branchless: out-of-range lanes clamp to row/col 0 (valid memory);
// their results only land in rows/cols discarded by the store guards.
// ---------------------------------------------------------------------------
__global__ void gemm_wmma_kernel(const float* __restrict__ A,
                                 const _Float16* __restrict__ WT,
                                 const float* __restrict__ bias,
                                 float* __restrict__ C,
                                 int M, int N, int K, int relu) {
  const int lane = threadIdx.x & 31;
  const int lo = lane & 15;
  const int hi = lane >> 4;
  const int n0 = blockIdx.x * 32 + lo;
  const int n1 = n0 + 16;
  const int m  = blockIdx.y * 16 + lo;

  const float*    arow = A  + (size_t)(m  < M ? m  : 0) * K;
  const _Float16* b0p  = WT + (size_t)(n0 < N ? n0 : 0) * K + hi * 16;
  const _Float16* b1p  = WT + (size_t)(n1 < N ? n1 : 0) * K + hi * 16;

  v8f acc0 = {}, acc1 = {};
  for (int kc = 0; kc < K; kc += 32) {
    const float* p = arow + kc + hi * 16;
    float4 a0 = *(const float4*)(p);
    float4 a1 = *(const float4*)(p + 4);
    float4 a2 = *(const float4*)(p + 8);
    float4 a3 = *(const float4*)(p + 12);
    v16h a;
    a[0]  = (_Float16)a0.x; a[1]  = (_Float16)a0.y;
    a[2]  = (_Float16)a0.z; a[3]  = (_Float16)a0.w;
    a[4]  = (_Float16)a1.x; a[5]  = (_Float16)a1.y;
    a[6]  = (_Float16)a1.z; a[7]  = (_Float16)a1.w;
    a[8]  = (_Float16)a2.x; a[9]  = (_Float16)a2.y;
    a[10] = (_Float16)a2.z; a[11] = (_Float16)a2.w;
    a[12] = (_Float16)a3.x; a[13] = (_Float16)a3.y;
    a[14] = (_Float16)a3.z; a[15] = (_Float16)a3.w;

    v16h bb0 = *(const v16h*)(b0p + kc);
    v16h bb1 = *(const v16h*)(b1p + kc);

    // stream-ahead hint on the (streamed-once) weight columns; speculative
    // prefetch past the end of the buffer is silently dropped.
    __builtin_prefetch(b0p + kc + 256, 0, 1);
    __builtin_prefetch(b1p + kc + 256, 0, 1);

    acc0 = __builtin_amdgcn_wmma_f32_16x16x32_f16(
        false, a, false, bb0, (short)0, acc0, false, false);
    acc1 = __builtin_amdgcn_wmma_f32_16x16x32_f16(
        false, a, false, bb1, (short)0, acc1, false, false);
  }

  const int mbase = blockIdx.y * 16 + hi * 8;
  if (n0 < N) {
    float bv = bias ? bias[n0] : 0.0f;
#pragma unroll
    for (int r = 0; r < 8; ++r) {
      int mm = mbase + r;
      if (mm < M) {
        float val = acc0[r] + bv;
        if (relu) val = fmaxf(val, 0.0f);
        C[(size_t)mm * N + n0] = val;
      }
    }
  }
  if (n1 < N) {
    float bv = bias ? bias[n1] : 0.0f;
#pragma unroll
    for (int r = 0; r < 8; ++r) {
      int mm = mbase + r;
      if (mm < M) {
        float val = acc1[r] + bv;
        if (relu) val = fmaxf(val, 0.0f);
        C[(size_t)mm * N + n1] = val;
      }
    }
  }
}

// ---------------------------------------------------------------------------
// Encoder embedding: x[s*B+b, e] = emb[tok[b,s], e] + sinusoidal PE(s, e)
// ---------------------------------------------------------------------------
__global__ void embed_enc_kernel(const int* __restrict__ toks,
                                 const float* __restrict__ emb,
                                 float* __restrict__ x) {
  int idx = blockIdx.x * blockDim.x + threadIdx.x;
  if (idx >= kTenc * kE) return;
  int e = idx % kE;
  int t = idx / kE;
  int b = t % kB;
  int s = t / kB;
  int tok = toks[b * kS + s];
  int i2 = e & ~1;
  float div = __expf(-9.210340371976184f * (float)i2 / (float)kE);
  float ang = (float)s * div;
  float pe = (e & 1) ? __cosf(ang) : __sinf(ang);
  x[idx] = emb[(size_t)tok * kE + e] + pe;
}

// Decoder embedding: no PE, tokens laid out [Lbuf*B] with t = l*B + b
__global__ void embed_dec_kernel(const int* __restrict__ toks,
                                 const float* __restrict__ emb,
                                 float* __restrict__ y) {
  int idx = blockIdx.x * blockDim.x + threadIdx.x;
  if (idx >= kTdec * kE) return;
  int e = idx % kE;
  int t = idx / kE;
  int tok = toks[t];
  y[idx] = emb[(size_t)tok * kE + e];
}

__global__ void init_tokens_kernel(int* __restrict__ toks) {
  int idx = blockIdx.x * blockDim.x + threadIdx.x;
  if (idx < kLbuf * kB) toks[idx] = kStart;
}

// ---------------------------------------------------------------------------
// LayerNorm with optional residual: out = LN(in + res) * g + b   (row = block)
// ---------------------------------------------------------------------------
__global__ void ln_kernel(const float* __restrict__ in,
                          const float* __restrict__ res,
                          const float* __restrict__ g,
                          const float* __restrict__ bta,
                          float* __restrict__ out, int E) {
  int row = blockIdx.x;
  int tid = threadIdx.x;
  const float* ip = in + (size_t)row * E;
  const float* rp = res ? res + (size_t)row * E : nullptr;
  float s = 0.0f, sq = 0.0f;
  for (int e = tid; e < E; e += blockDim.x) {
    float v = ip[e] + (rp ? rp[e] : 0.0f);
    s += v; sq += v * v;
  }
  __shared__ float rs[256], rq[256];
  rs[tid] = s; rq[tid] = sq;
  __syncthreads();
  for (int st = 128; st > 0; st >>= 1) {
    if (tid < st) { rs[tid] += rs[tid + st]; rq[tid] += rq[tid + st]; }
    __syncthreads();
  }
  float mean = rs[0] / (float)E;
  float var = rq[0] / (float)E - mean * mean;
  float rstd = rsqrtf(var + 1e-5f);
  float* op = out + (size_t)row * E;
  for (int e = tid; e < E; e += blockDim.x) {
    float v = ip[e] + (rp ? rp[e] : 0.0f);
    op[e] = (v - mean) * rstd * g[e] + bta[e];
  }
}

// ---------------------------------------------------------------------------
// MHA core: per (qpos, b, h) block of 64 threads. q/k/v rows t = pos*B + b,
// head slice cols [h*64, h*64+64). Scores + softmax + weighted V.
// ---------------------------------------------------------------------------
__global__ void attn_kernel(const float* __restrict__ q, int ldq,
                            const float* __restrict__ k, int ldk,
                            const float* __restrict__ v, int ldv,
                            float* __restrict__ o, int ldo,
                            int Bn, int Lk, int causal, float scale) {
  int qpos = blockIdx.x;
  int bh = blockIdx.y;
  int h = bh % kH;
  int b = bh / kH;
  int tid = threadIdx.x;  // 0..63
  __shared__ float s[64];
  __shared__ float sinv;
  const float* qrow = q + (size_t)(qpos * Bn + b) * ldq + h * kD;
  if (tid < Lk) {
    const float* krow = k + (size_t)(tid * Bn + b) * ldk + h * kD;
    float acc = 0.0f;
#pragma unroll 8
    for (int d = 0; d < kD; ++d) acc += qrow[d] * krow[d];
    acc *= scale;
    if (causal && tid > qpos) acc = -INFINITY;
    s[tid] = acc;
  }
  __syncthreads();
  if (tid == 0) {
    float mx = -INFINITY;
    for (int j = 0; j < Lk; ++j) mx = fmaxf(mx, s[j]);
    float sum = 0.0f;
    for (int j = 0; j < Lk; ++j) { float e = __expf(s[j] - mx); s[j] = e; sum += e; }
    sinv = 1.0f / sum;
  }
  __syncthreads();
  float inv = sinv;
  float acc = 0.0f;
  for (int j = 0; j < Lk; ++j)
    acc += s[j] * v[(size_t)(j * Bn + b) * ldv + h * kD + tid];
  o[(size_t)(qpos * Bn + b) * ldo + h * kD + tid] = acc * inv;
}

// ---------------------------------------------------------------------------
// Argmax over vocab row, first-index tie rule; writes next token + pred float
// ---------------------------------------------------------------------------
__global__ void argmax_kernel(const float* __restrict__ prob,
                              int* __restrict__ toks,
                              float* __restrict__ pred_out, int step) {
  int b = blockIdx.x;
  int tid = threadIdx.x;
  const float* row = prob + (size_t)b * kVT;
  float bv = -INFINITY; int bi = 0x7fffffff;
  for (int i = tid; i < kVT; i += blockDim.x) {
    float vv = row[i];
    if (vv > bv || (vv == bv && i < bi)) { bv = vv; bi = i; }
  }
  __shared__ float sv[256];
  __shared__ int si[256];
  sv[tid] = bv; si[tid] = bi;
  __syncthreads();
  for (int st = 128; st > 0; st >>= 1) {
    if (tid < st) {
      if (sv[tid + st] > sv[tid] ||
          (sv[tid + st] == sv[tid] && si[tid + st] < si[tid])) {
        sv[tid] = sv[tid + st]; si[tid] = si[tid + st];
      }
    }
    __syncthreads();
  }
  if (tid == 0) {
    int nxt = si[0];
    if (step + 1 < kLbuf) toks[(size_t)(step + 1) * kB + b] = nxt;
    pred_out[(size_t)step * kB + b] = (float)nxt;
  }
}

// ---------------------------------------------------------------------------
extern "C" void kernel_launch(void* const* d_in, const int* in_sizes, int n_in,
                              void* d_out, int out_size, void* d_ws, size_t ws_size,
                              hipStream_t stream) {
  (void)in_sizes; (void)n_in; (void)out_size; (void)ws_size;

  const int*   input       = (const int*)d_in[0];
  const float* emb         = (const float*)d_in[1];
  const float* enc_attn_w  = (const float*)d_in[2];
  const float* enc_attn_b  = (const float*)d_in[3];
  const float* enc_attn_ow = (const float*)d_in[4];
  const float* enc_attn_ob = (const float*)d_in[5];
  const float* enc_ln1_g   = (const float*)d_in[6];
  const float* enc_ln1_b   = (const float*)d_in[7];
  const float* enc_ln2_g   = (const float*)d_in[8];
  const float* enc_ln2_b   = (const float*)d_in[9];
  const float* enc_ffn_w1  = (const float*)d_in[10];
  const float* enc_ffn_b1  = (const float*)d_in[11];
  const float* enc_ffn_w2  = (const float*)d_in[12];
  const float* enc_ffn_b2  = (const float*)d_in[13];
  const float* enc_norm_g  = (const float*)d_in[14];
  const float* enc_norm_b  = (const float*)d_in[15];
  const float* dec_sa_w    = (const float*)d_in[16];
  const float* dec_sa_b    = (const float*)d_in[17];
  const float* dec_sa_ow   = (const float*)d_in[18];
  const float* dec_sa_ob   = (const float*)d_in[19];
  const float* dec_ca_w    = (const float*)d_in[20];
  const float* dec_ca_b    = (const float*)d_in[21];
  const float* dec_ca_ow   = (const float*)d_in[22];
  const float* dec_ca_ob   = (const float*)d_in[23];
  const float* dec_ln1_g   = (const float*)d_in[24];
  const float* dec_ln1_b   = (const float*)d_in[25];
  const float* dec_ln2_g   = (const float*)d_in[26];
  const float* dec_ln2_b   = (const float*)d_in[27];
  const float* dec_ln3_g   = (const float*)d_in[28];
  const float* dec_ln3_b   = (const float*)d_in[29];
  const float* dec_ffn_w1  = (const float*)d_in[30];
  const float* dec_ffn_b1  = (const float*)d_in[31];
  const float* dec_ffn_w2  = (const float*)d_in[32];
  const float* dec_ffn_b2  = (const float*)d_in[33];
  const float* dec_norm_g  = (const float*)d_in[34];
  const float* dec_norm_b  = (const float*)d_in[35];
  const float* voc_w       = (const float*)d_in[36];
  const float* voc_b       = (const float*)d_in[37];

  float* out_f = (float*)d_out;                     // [24*8] preds, then dists
  float* dists = out_f + (size_t)kMaxLen * kB;      // [24, 8, 32000]

  // -------- bump allocator over workspace --------
  size_t off = 0;
  auto alloc_bytes = [&](size_t bytes) -> void* {
    off = (off + 255) & ~(size_t)255;
    void* p = (char*)d_ws + off;
    off += bytes;
    return p;
  };
  auto allocF = [&](size_t n) { return (float*)alloc_bytes(n * sizeof(float)); };
  auto allocH = [&](size_t n) { return (_Float16*)alloc_bytes(n * sizeof(_Float16)); };

  // -------- weight conversion (fp32 -> f16 transposed [N,K]) --------
  auto cvt = [&](const float* w, int Nfull, int c0, int Kd, int Nd) -> _Float16* {
    _Float16* p = allocH((size_t)Nd * Kd);
    int total = Nd * Kd;
    cvtT_kernel<<<(total + 255) / 256, 256, 0, stream>>>(w, Nfull, c0, Kd, Nd, p);
    return p;
  };

  _Float16 *enc_wT[kLE], *enc_owT[kLE], *enc_f1T[kLE], *enc_f2T[kLE];
  for (int l = 0; l < kLE; ++l) {
    enc_wT[l]  = cvt(enc_attn_w  + (size_t)l * kE * 3 * kE, 3 * kE, 0, kE, 3 * kE);
    enc_owT[l] = cvt(enc_attn_ow + (size_t)l * kE * kE,     kE,     0, kE, kE);
    enc_f1T[l] = cvt(enc_ffn_w1  + (size_t)l * kE * kF,     kF,     0, kE, kF);
    enc_f2T[l] = cvt(enc_ffn_w2  + (size_t)l * kF * kE,     kE,     0, kF, kE);
  }
  _Float16 *sa_wT[kLD], *sa_owT[kLD], *ca_wT[kLD], *ca_owT[kLD], *d_f1T[kLD], *d_f2T[kLD];
  for (int l = 0; l < kLD; ++l) {
    sa_wT[l]  = cvt(dec_sa_w  + (size_t)l * kE * 3 * kE, 3 * kE, 0, kE, 3 * kE);
    sa_owT[l] = cvt(dec_sa_ow + (size_t)l * kE * kE,     kE,     0, kE, kE);
    ca_wT[l]  = cvt(dec_ca_w  + (size_t)l * kE * 3 * kE, 3 * kE, 0, kE, 3 * kE);
    ca_owT[l] = cvt(dec_ca_ow + (size_t)l * kE * kE,     kE,     0, kE, kE);
    d_f1T[l]  = cvt(dec_ffn_w1 + (size_t)l * kE * kF,    kF,     0, kE, kF);
    d_f2T[l]  = cvt(dec_ffn_w2 + (size_t)l * kF * kE,    kE,     0, kF, kE);
  }
  _Float16* voc_wT = cvt(voc_w, kVT, 0, kE, kVT);

  // -------- activation buffers --------
  float* x_enc   = allocF((size_t)kTenc * kE);
  float* qkv_e   = allocF((size_t)kTenc * 3 * kE);
  float* attn_oe = allocF((size_t)kTenc * kE);
  float* proj_e  = allocF((size_t)kTenc * kE);
  float* ffn1_e  = allocF((size_t)kTenc * kF);
  float* memory  = allocF((size_t)kTenc * kE);
  float* mem_k[kLD]; float* mem_v[kLD];
  for (int l = 0; l < kLD; ++l) {
    mem_k[l] = allocF((size_t)kTenc * kE);
    mem_v[l] = allocF((size_t)kTenc * kE);
  }
  float* y       = allocF((size_t)kTdec * kE);
  float* qkv_d   = allocF((size_t)kTdec * 3 * kE);
  float* attn_od = allocF((size_t)kTdec * kE);
  float* cq      = allocF((size_t)kTdec * kE);
  float* proj_d  = allocF((size_t)kTdec * kE);
  float* ffn1_d  = allocF((size_t)kTdec * kF);
  float* yfin    = allocF((size_t)kTdec * kE);
  int*   toks    = (int*)alloc_bytes((size_t)kLbuf * kB * sizeof(int));

  auto gemm = [&](const float* A, const _Float16* WT, const float* bias,
                  float* C, int M, int N, int K, bool relu) {
    dim3 g((N + 31) / 32, (M + 15) / 16);
    gemm_wmma_kernel<<<g, 32, 0, stream>>>(A, WT, bias, C, M, N, K, relu ? 1 : 0);
  };
  auto ln = [&](const float* in, const float* res, const float* g,
                const float* b, float* out, int T) {
    ln_kernel<<<T, 256, 0, stream>>>(in, res, g, b, out, kE);
  };
  auto attn = [&](const float* q, int ldq, const float* k, int ldk,
                  const float* v, int ldv, float* o, int Lq, int Lk, int causal) {
    attn_kernel<<<dim3(Lq, kB * kH), 64, 0, stream>>>(
        q, ldq, k, ldk, v, ldv, o, kE, kB, Lk, causal, 0.125f);
  };

  // ================= Encoder =================
  embed_enc_kernel<<<(kTenc * kE + 255) / 256, 256, 0, stream>>>(input, emb, x_enc);
  for (int l = 0; l < kLE; ++l) {
    gemm(x_enc, enc_wT[l], enc_attn_b + (size_t)l * 3 * kE, qkv_e, kTenc, 3 * kE, kE, false);
    attn(qkv_e, 3 * kE, qkv_e + kE, 3 * kE, qkv_e + 2 * kE, 3 * kE, attn_oe, kS, kS, 0);
    gemm(attn_oe, enc_owT[l], enc_attn_ob + (size_t)l * kE, proj_e, kTenc, kE, kE, false);
    ln(x_enc, proj_e, enc_ln1_g + (size_t)l * kE, enc_ln1_b + (size_t)l * kE, x_enc, kTenc);
    gemm(x_enc, enc_f1T[l], enc_ffn_b1 + (size_t)l * kF, ffn1_e, kTenc, kF, kE, true);
    gemm(ffn1_e, enc_f2T[l], enc_ffn_b2 + (size_t)l * kE, proj_e, kTenc, kE, kF, false);
    ln(x_enc, proj_e, enc_ln2_g + (size_t)l * kE, enc_ln2_b + (size_t)l * kE, x_enc, kTenc);
  }
  ln(x_enc, nullptr, enc_norm_g, enc_norm_b, memory, kTenc);

  // Cross-attention K/V precompute per decoder layer (memory is fixed)
  for (int l = 0; l < kLD; ++l) {
    gemm(memory, ca_wT[l] + (size_t)kE * kE,     dec_ca_b + (size_t)l * 3 * kE + kE,
         mem_k[l], kTenc, kE, kE, false);
    gemm(memory, ca_wT[l] + (size_t)2 * kE * kE, dec_ca_b + (size_t)l * 3 * kE + 2 * kE,
         mem_v[l], kTenc, kE, kE, false);
  }

  // ================= Greedy decode =================
  init_tokens_kernel<<<1, 256, 0, stream>>>(toks);
  for (int i = 0; i < kMaxLen; ++i) {
    embed_dec_kernel<<<(kTdec * kE + 255) / 256, 256, 0, stream>>>(toks, emb, y);
    for (int l = 0; l < kLD; ++l) {
      // self-attention (causal)
      gemm(y, sa_wT[l], dec_sa_b + (size_t)l * 3 * kE, qkv_d, kTdec, 3 * kE, kE, false);
      attn(qkv_d, 3 * kE, qkv_d + kE, 3 * kE, qkv_d + 2 * kE, 3 * kE, attn_od, kLbuf, kLbuf, 1);
      gemm(attn_od, sa_owT[l], dec_sa_ob + (size_t)l * kE, proj_d, kTdec, kE, kE, false);
      ln(y, proj_d, dec_ln1_g + (size_t)l * kE, dec_ln1_b + (size_t)l * kE, y, kTdec);
      // cross-attention
      gemm(y, ca_wT[l], dec_ca_b + (size_t)l * 3 * kE, cq, kTdec, kE, kE, false);
      attn(cq, kE, mem_k[l], kE, mem_v[l], kE, attn_od, kLbuf, kS, 0);
      gemm(attn_od, ca_owT[l], dec_ca_ob + (size_t)l * kE, proj_d, kTdec, kE, kE, false);
      ln(y, proj_d, dec_ln2_g + (size_t)l * kE, dec_ln2_b + (size_t)l * kE, y, kTdec);
      // FFN
      gemm(y, d_f1T[l], dec_ffn_b1 + (size_t)l * kF, ffn1_d, kTdec, kF, kE, true);
      gemm(ffn1_d, d_f2T[l], dec_ffn_b2 + (size_t)l * kE, proj_d, kTdec, kE, kF, false);
      ln(y, proj_d, dec_ln3_g + (size_t)l * kE, dec_ln3_b + (size_t)l * kE, y, kTdec);
    }
    ln(y, nullptr, dec_norm_g, dec_norm_b, yfin, kTdec);
    // vocab projection for row i only: A = yfin rows [i*B, i*B+B)
    float* prob = dists + (size_t)i * kB * kVT;
    gemm(yfin + (size_t)i * kB * kE, voc_wT, voc_b, prob, kB, kVT, kE, false);
    argmax_kernel<<<kB, 256, 0, stream>>>(prob, toks, out_f, i);
  }
}
